// GATClassification_16896401342688
// MI455X (gfx1250) — matrix operand order repaired and checked by reading, not compiled, not run on hardware
//
#include <hip/hip_runtime.h>

typedef __attribute__((ext_vector_type(16))) __bf16 v16bf;
typedef __attribute__((ext_vector_type(8)))  float  v8f;

__device__ __forceinline__ float lrelu(float x, float s) { return x > 0.f ? x : s * x; }

// order-preserving float <-> uint for atomic max
__device__ __forceinline__ unsigned ordf(float f) {
  unsigned u = __float_as_uint(f);
  return (u & 0x80000000u) ? ~u : (u | 0x80000000u);
}
__device__ __forceinline__ float deord(unsigned u) {
  unsigned v = (u & 0x80000000u) ? (u ^ 0x80000000u) : ~u;
  return __uint_as_float(v);
}

__device__ __forceinline__ void edge_sd(const long long* ei, long long E, long long i,
                                        int& s, int& d) {
  if (i < E) { s = (int)ei[i]; d = (int)ei[E + i]; }
  else       { s = d = (int)(i - E); }                    // self loops appended
}

constexpr int RS = 136;  // bf16 row stride of pre-transposed weights (16B aligned, bank-spread)

// ---------------------------------------------------------------------------
// One-time weight prep: W[128][outc] fp32 -> Wbf[nc][RS] bf16, transposed so a
// B fragment (two runs of 8 K, 16 apart, for one column) is two 16B LDS reads.
// ---------------------------------------------------------------------------
__global__ __launch_bounds__(256) void k_convW(const float* __restrict__ W,
                                               __bf16* __restrict__ Wbf,
                                               int outc, int nc) {
  int idx = blockIdx.x * 256 + threadIdx.x;
  if (idx >= nc * RS) return;
  int ncol = idx / RS, k = idx % RS;
  float v = (k < 128 && ncol < outc) ? W[k * outc + ncol] : 0.f;
  Wbf[idx] = (__bf16)v;
}

// ---------------------------------------------------------------------------
// C[n,OUTC] = A[n,128] @ W[128,OUTC] (+ bias), bf16 WMMA, fp32 accumulate.
// Block: 256 threads = 8 waves; block covers 128 rows; wave covers 16 rows.
// Pre-formatted bf16 W is async-DMA'd into LDS (ASYNCcnt path). All four A
// K-fragments are loaded/converted up front so the 16 global_load_b128 stay
// pipelined instead of serializing against the accumulator registers.
// ---------------------------------------------------------------------------
template <int OUTC, bool BIAS>
__global__ __launch_bounds__(256) void k_gemm(const float* __restrict__ A,
                                              const __bf16* __restrict__ Wbf,
                                              const float* __restrict__ bias,
                                              float* __restrict__ C, int n) {
  constexpr int COLT = (OUTC + 15) / 16;
  constexpr int NC   = COLT * 16;
  constexpr int CHUNKS = NC * RS * 2 / 16;        // 16-byte chunks
  __shared__ __align__(16) __bf16 Wl[NC * RS];

  const int tid = threadIdx.x;
  // async copy global (pre-formatted bf16) -> LDS, 16B per lane per issue
  for (int idx = tid; idx < CHUNKS; idx += 256) {
    unsigned lds = (unsigned)(size_t)&Wl[idx * 8];
    const __bf16* g = Wbf + idx * 8;
    asm volatile("global_load_async_to_lds_b128 %0, %1, off"
                 :: "v"(lds), "v"(g) : "memory");
  }
  asm volatile("s_wait_asynccnt 0" ::: "memory");
  __syncthreads();

  const int lane = tid & 31, wave = tid >> 5;
  const int m  = lane & 15;                       // A row in frag / B,D column
  const int kg = lane >> 4;                       // lane half
  const int rowBase = blockIdx.x * 128 + wave * 16;
  const int row  = rowBase + m;
  const long long rowc = (row < n) ? row : (n - 1);

  // ---- load the full A row strip for this lane: 4 K-fragments ----
  v16bf af[4];
#pragma unroll
  for (int kc = 0; kc < 4; ++kc) {
    const float* ar = A + rowc * 128 + kc * 32 + kg * 8;
    float4 x0 = *reinterpret_cast<const float4*>(ar);
    float4 x1 = *reinterpret_cast<const float4*>(ar + 4);
    float4 x2 = *reinterpret_cast<const float4*>(ar + 16);
    float4 x3 = *reinterpret_cast<const float4*>(ar + 20);
    v16bf a;
    a[0] = (__bf16)x0.x; a[1] = (__bf16)x0.y; a[2]  = (__bf16)x0.z; a[3]  = (__bf16)x0.w;
    a[4] = (__bf16)x1.x; a[5] = (__bf16)x1.y; a[6]  = (__bf16)x1.z; a[7]  = (__bf16)x1.w;
    a[8] = (__bf16)x2.x; a[9] = (__bf16)x2.y; a[10] = (__bf16)x2.z; a[11] = (__bf16)x2.w;
    a[12]= (__bf16)x3.x; a[13]= (__bf16)x3.y; a[14] = (__bf16)x3.z; a[15] = (__bf16)x3.w;
    af[kc] = a;
  }

  v8f acc[COLT];
#pragma unroll
  for (int t = 0; t < COLT; ++t) acc[t] = (v8f){0, 0, 0, 0, 0, 0, 0, 0};

#pragma unroll
  for (int kc = 0; kc < 4; ++kc) {                // K = 4 x 32
#pragma unroll
    for (int t = 0; t < COLT; ++t) {
      const __bf16* bp = &Wl[(t * 16 + m) * RS + kc * 32 + kg * 8];
      union { uint4 q[2]; v16bf v; } bu;
      bu.q[0] = *reinterpret_cast<const uint4*>(bp);
      bu.q[1] = *reinterpret_cast<const uint4*>(bp + 16);
      acc[t] = __builtin_amdgcn_wmma_f32_16x16x32_bf16(
          false, af[kc], false, bu.v, (short)0, acc[t], false, false);
    }
  }

#pragma unroll
  for (int t = 0; t < COLT; ++t) {
#pragma unroll
    for (int v = 0; v < 8; ++v) {
      int r = rowBase + kg * 8 + v;               // D layout: M = (lane/16)*8 + v
      int c = t * 16 + m;                         // N = lane%16
      if (r < n && (OUTC % 16 == 0 || c < OUTC)) {
        float o = acc[t][v];
        if (BIAS) o += bias[c];
        C[(long long)r * OUTC + c] = o;
      }
    }
  }
}

// per-(node,head) attention logits: al = sum_c h[node,head,c] * a[head,c]
__global__ __launch_bounds__(256) void k_attlogits(const float* __restrict__ h,
                                                   const float* __restrict__ a_src,
                                                   const float* __restrict__ a_dst,
                                                   float* __restrict__ als,
                                                   float* __restrict__ ald, int n) {
  int t = blockIdx.x * blockDim.x + threadIdx.x;
  if (t >= n * 4) return;
  int node = t >> 2, head = t & 3;
  const float4* hp = reinterpret_cast<const float4*>(h + (long long)node * 128 + head * 32);
  const float4* as = reinterpret_cast<const float4*>(a_src + head * 32);
  const float4* ad = reinterpret_cast<const float4*>(a_dst + head * 32);
  float s = 0.f, d = 0.f;
#pragma unroll
  for (int j = 0; j < 8; ++j) {
    float4 hv = hp[j], av = as[j], bv = ad[j];
    s += hv.x * av.x + hv.y * av.y + hv.z * av.z + hv.w * av.w;
    d += hv.x * bv.x + hv.y * bv.y + hv.z * bv.z + hv.w * bv.w;
  }
  als[t] = s; ald[t] = d;
}

__global__ __launch_bounds__(256) void k_init(float* __restrict__ acc,
                                              unsigned* __restrict__ mord,
                                              float* __restrict__ denom, int n) {
  long long idx = (long long)blockIdx.x * blockDim.x + threadIdx.x;
  if (idx < (long long)n * 128) acc[idx] = 0.f;
  if (idx < (long long)n * 4) { mord[idx] = 0u; denom[idx] = 0.f; }
}

__global__ __launch_bounds__(256) void k_edge_max(const long long* __restrict__ ei,
                                                  long long E, int n,
                                                  const float* __restrict__ als,
                                                  const float* __restrict__ ald,
                                                  unsigned* __restrict__ mord) {
  long long i = (long long)blockIdx.x * blockDim.x + threadIdx.x;
  if (i >= E + n) return;
  int s, d; edge_sd(ei, E, i, s, d);
  float4 a = *reinterpret_cast<const float4*>(als + (long long)s * 4);
  float4 b = *reinterpret_cast<const float4*>(ald + (long long)d * 4);
  unsigned* mp = mord + (long long)d * 4;
  atomicMax(mp + 0, ordf(lrelu(a.x + b.x, 0.2f)));
  atomicMax(mp + 1, ordf(lrelu(a.y + b.y, 0.2f)));
  atomicMax(mp + 2, ordf(lrelu(a.z + b.z, 0.2f)));
  atomicMax(mp + 3, ordf(lrelu(a.w + b.w, 0.2f)));
}

__global__ __launch_bounds__(256) void k_edge_sum(const long long* __restrict__ ei,
                                                  long long E, int n,
                                                  const float* __restrict__ als,
                                                  const float* __restrict__ ald,
                                                  const unsigned* __restrict__ mord,
                                                  float* __restrict__ denom) {
  long long i = (long long)blockIdx.x * blockDim.x + threadIdx.x;
  if (i >= E + n) return;
  int s, d; edge_sd(ei, E, i, s, d);
  float4 a = *reinterpret_cast<const float4*>(als + (long long)s * 4);
  float4 b = *reinterpret_cast<const float4*>(ald + (long long)d * 4);
  const unsigned* mp = mord + (long long)d * 4;
  float* dp = denom + (long long)d * 4;
  unsafeAtomicAdd(dp + 0, __expf(lrelu(a.x + b.x, 0.2f) - deord(mp[0])));
  unsafeAtomicAdd(dp + 1, __expf(lrelu(a.y + b.y, 0.2f) - deord(mp[1])));
  unsafeAtomicAdd(dp + 2, __expf(lrelu(a.z + b.z, 0.2f) - deord(mp[2])));
  unsafeAtomicAdd(dp + 3, __expf(lrelu(a.w + b.w, 0.2f) - deord(mp[3])));
}

// one wave per edge: lane handles 4 channels (head = lane/8)
__global__ __launch_bounds__(256) void k_edge_aggr(const long long* __restrict__ ei,
                                                   long long E, int n,
                                                   const float* __restrict__ als,
                                                   const float* __restrict__ ald,
                                                   const unsigned* __restrict__ mord,
                                                   const float* __restrict__ denom,
                                                   const float* __restrict__ h,
                                                   float* __restrict__ acc) {
  long long i = (long long)blockIdx.x * 8 + (threadIdx.x >> 5);
  if (i >= E + n) return;
  int lane = threadIdx.x & 31;
  int s, d; edge_sd(ei, E, i, s, d);
  int head = lane >> 3;
  float e = lrelu(als[(long long)s * 4 + head] + ald[(long long)d * 4 + head], 0.2f);
  float mx = deord(mord[(long long)d * 4 + head]);
  float alpha = __expf(e - mx) / (denom[(long long)d * 4 + head] + 1e-16f);
  int c0 = lane * 4;
  float4 hv = *reinterpret_cast<const float4*>(h + (long long)s * 128 + c0);
  float* ap = acc + (long long)d * 128 + c0;
  unsafeAtomicAdd(ap + 0, hv.x * alpha);
  unsafeAtomicAdd(ap + 1, hv.y * alpha);
  unsafeAtomicAdd(ap + 2, hv.z * alpha);
  unsafeAtomicAdd(ap + 3, hv.w * alpha);
}

__global__ __launch_bounds__(256) void k_bias_act(const float* __restrict__ acc,
                                                  const float* __restrict__ b,
                                                  float* __restrict__ out, int n) {
  long long idx = (long long)blockIdx.x * blockDim.x + threadIdx.x;   // float4 units
  if (idx >= (long long)n * 32) return;
  int c = ((int)(idx & 31)) * 4;
  float4 v = reinterpret_cast<const float4*>(acc)[idx];
  float4 bv = *reinterpret_cast<const float4*>(b + c);
  v.x = lrelu(v.x + bv.x, 0.01f);
  v.y = lrelu(v.y + bv.y, 0.01f);
  v.z = lrelu(v.z + bv.z, 0.01f);
  v.w = lrelu(v.w + bv.w, 0.01f);
  reinterpret_cast<float4*>(out)[idx] = v;
}

extern "C" void kernel_launch(void* const* d_in, const int* in_sizes, int n_in,
                              void* d_out, int out_size, void* d_ws, size_t ws_size,
                              hipStream_t stream) {
  const float*     x   = (const float*)d_in[0];
  const long long* ei  = (const long long*)d_in[1];
  const float*     W1  = (const float*)d_in[2];
  const float*     a1s = (const float*)d_in[3];
  const float*     a1d = (const float*)d_in[4];
  const float*     b1  = (const float*)d_in[5];
  const float*     W2  = (const float*)d_in[6];
  const float*     a2s = (const float*)d_in[7];
  const float*     a2d = (const float*)d_in[8];
  const float*     b2  = (const float*)d_in[9];
  const float*     Wc  = (const float*)d_in[10];
  const float*     bc  = (const float*)d_in[11];
  float* out = (float*)d_out;

  const int N = in_sizes[0] / 128;
  const long long E = in_sizes[1] / 2;
  const long long ET = E + N;

  float* ws = (float*)d_ws;
  float*    h     = ws;                        // N*128
  float*    acc   = h + (long long)N * 128;    // N*128
  float*    hact  = acc + (long long)N * 128;  // N*128
  float*    als   = hact + (long long)N * 128; // N*4
  float*    ald   = als + (long long)N * 4;    // N*4
  unsigned* mord  = (unsigned*)(ald + (long long)N * 4); // N*4
  float*    denom = (float*)(mord + (long long)N * 4);   // N*4
  __bf16*   w1bf  = (__bf16*)(denom + (long long)N * 4); // 128*RS
  __bf16*   w2bf  = w1bf + 128 * RS;                     // 128*RS
  __bf16*   wcbf  = w2bf + 128 * RS;                     // 32*RS

  dim3 blk(256);
  dim3 gGemm((N + 127) / 128);
  dim3 gAtt((N * 4 + 255) / 256);
  dim3 gInit((unsigned)(((long long)N * 128 + 255) / 256));
  dim3 gEdge((unsigned)((ET + 255) / 256));
  dim3 gAggr((unsigned)((ET + 7) / 8));
  dim3 gAct((unsigned)(((long long)N * 32 + 255) / 256));
  dim3 gCw128((128 * RS + 255) / 256);
  dim3 gCw32((32 * RS + 255) / 256);

  // one-time weight prep (bf16, transposed, padded to LDS layout)
  k_convW<<<gCw128, blk, 0, stream>>>(W1, w1bf, 128, 128);
  k_convW<<<gCw128, blk, 0, stream>>>(W2, w2bf, 128, 128);
  k_convW<<<gCw32,  blk, 0, stream>>>(Wc, wcbf, 20, 32);

  // ---- layer 1 ----
  k_gemm<128, false><<<gGemm, blk, 0, stream>>>(x, w1bf, nullptr, h, N);
  k_attlogits<<<gAtt, blk, 0, stream>>>(h, a1s, a1d, als, ald, N);
  k_init<<<gInit, blk, 0, stream>>>(acc, mord, denom, N);
  k_edge_max<<<gEdge, blk, 0, stream>>>(ei, E, N, als, ald, mord);
  k_edge_sum<<<gEdge, blk, 0, stream>>>(ei, E, N, als, ald, mord, denom);
  k_edge_aggr<<<gAggr, blk, 0, stream>>>(ei, E, N, als, ald, mord, denom, h, acc);
  k_bias_act<<<gAct, blk, 0, stream>>>(acc, b1, hact, N);

  // ---- layer 2 ----
  k_gemm<128, false><<<gGemm, blk, 0, stream>>>(hact, w2bf, nullptr, h, N);
  k_attlogits<<<gAtt, blk, 0, stream>>>(h, a2s, a2d, als, ald, N);
  k_init<<<gInit, blk, 0, stream>>>(acc, mord, denom, N);
  k_edge_max<<<gEdge, blk, 0, stream>>>(ei, E, N, als, ald, mord);
  k_edge_sum<<<gEdge, blk, 0, stream>>>(ei, E, N, als, ald, mord, denom);
  k_edge_aggr<<<gAggr, blk, 0, stream>>>(ei, E, N, als, ald, mord, denom, h, acc);
  k_bias_act<<<gAct, blk, 0, stream>>>(acc, b2, hact, N);

  // ---- classifier ----
  k_gemm<20, true><<<gGemm, blk, 0, stream>>>(hact, wcbf, bc, out, N);
}
